// DeformableTransformer_73143293051450
// MI455X (gfx1250) — matrix operand order repaired and checked by reading, not compiled, not run on hardware
//
#include <hip/hip_runtime.h>

// ---------------------------------------------------------------------------
// Deformable transformer (DETR-style) for MI455X / gfx1250, wave32 + WMMA bf16
// GEMM: 32x64 supertile per wave, 8 v_wmma_f32_16x16x32_bf16 per K-step.
// ---------------------------------------------------------------------------

#define CDIV(a, b) (((a) + (b) - 1) / (b))

typedef __attribute__((ext_vector_type(16))) __bf16 v16bf;
typedef __attribute__((ext_vector_type(8)))  float  v8f;
typedef unsigned int u32x4 __attribute__((ext_vector_type(4)));
typedef unsigned short u16;

__device__ __forceinline__ u16 f2bf(float f) {
  unsigned u = __float_as_uint(f);
  u += 0x7FFFu + ((u >> 16) & 1u);           // round-to-nearest-even
  return (u16)(u >> 16);
}

union Frag { v16bf v; u32x4 q[2]; };

// ------------------------- WMMA GEMM (bf16 -> f32) -------------------------
// C[M,N] = A[M,K] (bf16 row-major) * Bp (bf16, packed fragments) + bias
// One wave computes a 32(M) x 64(N) supertile: 2 A-frags x 4 B-frags -> 8 acc.
// Packed B layout: [nts][kt][j=0..3][lane=0..31][16 bf16]  (4KB per (nts,kt))
__global__ __launch_bounds__(256) void gemm_bf16_wmma(
    const u16* __restrict__ A, const u16* __restrict__ Bp,
    const float* __restrict__ bias, float* __restrict__ Cf,
    u16* __restrict__ Cb, int M, int N, int K, int relu) {
  int lane   = threadIdx.x & 31;
  int tile   = blockIdx.x * 8 + (threadIdx.x >> 5);
  int tilesN = N >> 6;                         // 64-col supertiles
  int tilesM = (M + 31) >> 5;                  // 32-row supertiles
  if (tile >= tilesM * tilesN) return;
  int mt = tile / tilesN;
  int nt = tile - mt * tilesN;

  int r0 = mt * 32 + (lane & 15);
  int r1 = r0 + 16;
  if (r0 > M - 1) r0 = M - 1;                  // clamp; masked on store
  if (r1 > M - 1) r1 = M - 1;
  // A fragment: lane<16 holds K {0..7,16..23}, lane>=16 holds K {8..15,24..31}
  const u16* arow0 = A + (size_t)r0 * K + ((lane >> 4) << 3);
  const u16* arow1 = A + (size_t)r1 * K + ((lane >> 4) << 3);
  const u16* bptr  = Bp + ((size_t)nt * (K >> 5)) * 2048 + (lane << 4);

  v8f acc[2][4];
#pragma unroll
  for (int h = 0; h < 2; ++h)
#pragma unroll
    for (int j = 0; j < 4; ++j)
      acc[h][j] = (v8f){0.f, 0.f, 0.f, 0.f, 0.f, 0.f, 0.f, 0.f};

  for (int kb = 0; kb < K; kb += 32) {
    Frag a0, a1;
    a0.q[0] = *(const u32x4*)(arow0 + kb);
    a0.q[1] = *(const u32x4*)(arow0 + kb + 16);
    a1.q[0] = *(const u32x4*)(arow1 + kb);
    a1.q[1] = *(const u32x4*)(arow1 + kb + 16);
#pragma unroll
    for (int j = 0; j < 4; ++j) {
      Frag b;
      b.q[0] = *(const u32x4*)(bptr + j * 512);
      b.q[1] = *(const u32x4*)(bptr + j * 512 + 8);
      acc[0][j] = __builtin_amdgcn_wmma_f32_16x16x32_bf16(
          false, a0.v, false, b.v, (short)0, acc[0][j], false, false);
      acc[1][j] = __builtin_amdgcn_wmma_f32_16x16x32_bf16(
          false, a1.v, false, b.v, (short)0, acc[1][j], false, false);
    }
    bptr += 2048;
  }

#pragma unroll
  for (int j = 0; j < 4; ++j) {
    int col = nt * 64 + j * 16 + (lane & 15);
    float bv = bias ? bias[col] : 0.f;
#pragma unroll
    for (int h = 0; h < 2; ++h) {
      int mbase = mt * 32 + h * 16 + ((lane >> 4) << 3);
#pragma unroll
      for (int r = 0; r < 8; ++r) {
        int m = mbase + r;
        if (m < M) {
          float v = acc[h][j][r] + bv;
          if (relu) v = fmaxf(v, 0.f);
          if (Cf) Cf[(size_t)m * N + col] = v;
          if (Cb) Cb[(size_t)m * N + col] = f2bf(v);
        }
      }
    }
  }
}

// ---------------- pack fp32 weight [K,Nfull] -> WMMA B fragments -----------
// layout: [nts][kt][j][lane][16]
__global__ __launch_bounds__(256) void pack_kernel(
    const float* __restrict__ W, u16* __restrict__ Bp,
    int K, int Nfull, int n0, int ncols) {
  int e = blockIdx.x * 256 + threadIdx.x;
  int tilesK = K >> 5;
  int total = (ncols >> 6) * tilesK * 2048;
  if (e >= total) return;
  int i    = e & 15;
  int lane = (e >> 4) & 31;
  int j    = (e >> 9) & 3;
  int rest = e >> 11;
  int kt   = rest % tilesK;
  int nts  = rest / tilesK;
  int k    = kt * 32 + ((i >> 3) << 4) + ((lane >> 4) << 3) + (i & 7);    // lane<16: K {0..7,16..23}, else {8..15,24..31}
  int col  = n0 + nts * 64 + j * 16 + (lane & 15);
  Bp[e] = f2bf(W[(size_t)k * Nfull + col]);
}

// ---------------- flatten src/pos levels, add level embed ------------------
__global__ __launch_bounds__(256) void flatten_kernel(
    const float* __restrict__ s0, const float* __restrict__ s1,
    const float* __restrict__ s2, const float* __restrict__ s3,
    const float* __restrict__ p0, const float* __restrict__ p1,
    const float* __restrict__ p2, const float* __restrict__ p3,
    const float* __restrict__ lvl_emb,
    float* __restrict__ xf, u16* __restrict__ xbf, float* __restrict__ pf) {
  int idx = blockIdx.x * 256 + threadIdx.x;
  const int total = 26588 * 256;
  if (idx >= total) return;
  int c  = idx & 255;
  int ts = idx >> 8;               // b*13294 + s
  int b  = ts / 13294;
  int s  = ts - b * 13294;
  int lvl, sl, HW;
  const float *sp, *pp;
  if (s < 10000)      { lvl = 0; sl = s;         HW = 10000; sp = s0; pp = p0; }
  else if (s < 12500) { lvl = 1; sl = s - 10000; HW = 2500;  sp = s1; pp = p1; }
  else if (s < 13125) { lvl = 2; sl = s - 12500; HW = 625;   sp = s2; pp = p2; }
  else                { lvl = 3; sl = s - 13125; HW = 169;   sp = s3; pp = p3; }
  size_t sidx = ((size_t)b * 256 + c) * HW + sl;     // (B,C,H,W) layout
  float v = sp[sidx];
  xf[idx]  = v;
  xbf[idx] = f2bf(v);
  pf[idx]  = pp[sidx] + lvl_emb[lvl * 256 + c];
}

// ------------------------------- elementwise -------------------------------
__global__ __launch_bounds__(256) void addq_kernel(
    const float* __restrict__ a, const float* __restrict__ p,
    u16* __restrict__ o, int n) {
  int i = blockIdx.x * 256 + threadIdx.x;
  if (i < n) o[i] = f2bf(a[i] + p[i]);
}

__global__ __launch_bounds__(256) void dec_init_kernel(
    const float* __restrict__ qe, float* __restrict__ y, u16* __restrict__ ybf) {
  int i = blockIdx.x * 256 + threadIdx.x;
  if (i >= 600 * 256) return;
  int c = i & 255;
  int q = (i >> 8) % 300;
  float v = qe[(size_t)q * 512 + 256 + c];
  y[i] = v;
  ybf[i] = f2bf(v);
}

__global__ __launch_bounds__(256) void dec_addq_kernel(
    const float* __restrict__ y, const float* __restrict__ qe, u16* __restrict__ o) {
  int i = blockIdx.x * 256 + threadIdx.x;
  if (i >= 600 * 256) return;
  int c = i & 255;
  int q = (i >> 8) % 300;
  o[i] = f2bf(y[i] + qe[(size_t)q * 512 + c]);
}

// ----------------------- residual + layernorm (wave/token) -----------------
__global__ __launch_bounds__(256) void ln_kernel(
    const float* __restrict__ x, const float* __restrict__ d,
    const float* __restrict__ s, const float* __restrict__ bb,
    float* __restrict__ out, u16* __restrict__ out_bf, int ntok) {
  int lane = threadIdx.x & 31;
  int tok  = blockIdx.x * 8 + (threadIdx.x >> 5);
  if (tok >= ntok) return;
  const float* xr = x + (size_t)tok * 256;
  const float* dr = d + (size_t)tok * 256;
  float v[8];
  float sum = 0.f;
#pragma unroll
  for (int i = 0; i < 8; ++i) { v[i] = xr[lane + i * 32] + dr[lane + i * 32]; sum += v[i]; }
#pragma unroll
  for (int o = 16; o > 0; o >>= 1) sum += __shfl_xor(sum, o);
  float mean = sum * (1.f / 256.f);
  float var = 0.f;
#pragma unroll
  for (int i = 0; i < 8; ++i) { float t = v[i] - mean; var += t * t; }
#pragma unroll
  for (int o = 16; o > 0; o >>= 1) var += __shfl_xor(var, o);
  float inv = rsqrtf(var * (1.f / 256.f) + 1e-5f);
#pragma unroll
  for (int i = 0; i < 8; ++i) {
    int c = lane + i * 32;
    float ov = (v[i] - mean) * inv * s[c] + bb[c];
    out[(size_t)tok * 256 + c] = ov;
    if (out_bf) out_bf[(size_t)tok * 256 + c] = f2bf(ov);
  }
}

// ------------- multiscale deformable attention sampler (fused softmax) -----
// one wave per (b, token, head); lane == channel (DH == 32 == wave32)
__global__ __launch_bounds__(256) void msda_kernel(
    const float* __restrict__ value, const float* __restrict__ off,
    const float* __restrict__ awl, const float* __restrict__ refq,
    u16* __restrict__ out_bf, int ntok, int mode) {
  const int lvlW[4] = {100, 50, 25, 13};
  const int lvlH[4] = {100, 50, 25, 13};
  const int lvlS[4] = {0, 10000, 12500, 13125};
  int lane = threadIdx.x & 31;
  int wave = blockIdx.x * 8 + (threadIdx.x >> 5);
  if (wave >= ntok * 8) return;
  int t = wave >> 3;
  int h = wave & 7;
  int b;
  float rx, ry;
  if (mode == 0) {
    b = t / 13294;
    int s = t - b * 13294;
    int W, H, sl;
    if (s < 10000)      { W = 100; H = 100; sl = s; }
    else if (s < 12500) { W = 50;  H = 50;  sl = s - 10000; }
    else if (s < 13125) { W = 25;  H = 25;  sl = s - 12500; }
    else                { W = 13;  H = 13;  sl = s - 13125; }
    int yy = sl / W, xx = sl - yy * W;
    rx = (xx + 0.5f) / (float)W;
    ry = (yy + 0.5f) / (float)H;
  } else {
    b = t / 300;
    rx = refq[t * 2 + 0];
    ry = refq[t * 2 + 1];
  }
  // softmax over the 16 (L*P) attention logits for this head
  const float* ap = awl + (size_t)t * 128 + h * 16;
  float w[16], mx = -1e30f;
#pragma unroll
  for (int j = 0; j < 16; ++j) { w[j] = ap[j]; mx = fmaxf(mx, w[j]); }
  float sum = 0.f;
#pragma unroll
  for (int j = 0; j < 16; ++j) { w[j] = __expf(w[j] - mx); sum += w[j]; }
  float inv = 1.f / sum;

  const float* op = off + (size_t)t * 256 + h * 32;
  const float* vb = value + (size_t)b * 13294 * 256 + h * 32 + lane;
  float acc = 0.f;
#pragma unroll
  for (int l = 0; l < 4; ++l) {
    int W = lvlW[l], H = lvlH[l], ls = lvlS[l];
#pragma unroll
    for (int p = 0; p < 4; ++p) {
      float ox = op[l * 8 + p * 2];
      float oy = op[l * 8 + p * 2 + 1];
      float X = rx * (float)W + ox - 0.5f;
      float Y = ry * (float)H + oy - 0.5f;
      float x0f = floorf(X), y0f = floorf(Y);
      float fx = X - x0f, fy = Y - y0f;
      int x0 = (int)x0f, y0 = (int)y0f;
      float s00 = 0.f, s10 = 0.f, s01 = 0.f, s11 = 0.f;
      bool vx0 = (x0 >= 0) & (x0 < W);
      bool vx1 = (x0 + 1 >= 0) & (x0 + 1 < W);
      bool vy0 = (y0 >= 0) & (y0 < H);
      bool vy1 = (y0 + 1 >= 0) & (y0 + 1 < H);
      if (vx0 & vy0) s00 = vb[(size_t)(ls + y0 * W + x0) * 256];
      if (vx1 & vy0) s10 = vb[(size_t)(ls + y0 * W + x0 + 1) * 256];
      if (vx0 & vy1) s01 = vb[(size_t)(ls + (y0 + 1) * W + x0) * 256];
      if (vx1 & vy1) s11 = vb[(size_t)(ls + (y0 + 1) * W + x0 + 1) * 256];
      float bil = s00 * (1.f - fx) * (1.f - fy) + s10 * fx * (1.f - fy) +
                  s01 * (1.f - fx) * fy + s11 * fx * fy;
      acc += w[l * 4 + p] * inv * bil;
    }
  }
  out_bf[(size_t)t * 256 + h * 32 + lane] = f2bf(acc);
}

// --------------------------- decoder self-attention ------------------------
__global__ __launch_bounds__(256) void sa_scores_kernel(
    const float* __restrict__ qk, float* __restrict__ att) {
  int i = blockIdx.x * 256 + threadIdx.x;
  if (i >= 2 * 8 * 300 * 300) return;
  int k = i % 300;
  int r = i / 300;
  int q = r % 300; r /= 300;
  int h = r & 7;
  int b = r >> 3;
  const float* Q  = qk + ((size_t)(b * 300 + q)) * 512 + h * 32;
  const float* Kp = qk + ((size_t)(b * 300 + k)) * 512 + 256 + h * 32;
  float sv = 0.f;
#pragma unroll
  for (int d2 = 0; d2 < 32; ++d2) sv += Q[d2] * Kp[d2];
  att[i] = sv * 0.1767766953f;     // 1/sqrt(32)
}

__global__ __launch_bounds__(256) void sa_softmax_kernel(float* __restrict__ att) {
  int lane = threadIdx.x & 31;
  int row  = blockIdx.x * 8 + (threadIdx.x >> 5);
  if (row >= 4800) return;
  float* a = att + (size_t)row * 300;
  float mx = -1e30f;
  for (int i = lane; i < 300; i += 32) mx = fmaxf(mx, a[i]);
#pragma unroll
  for (int o = 16; o > 0; o >>= 1) mx = fmaxf(mx, __shfl_xor(mx, o));
  float sum = 0.f;
  for (int i = lane; i < 300; i += 32) { float e = __expf(a[i] - mx); a[i] = e; sum += e; }
#pragma unroll
  for (int o = 16; o > 0; o >>= 1) sum += __shfl_xor(sum, o);
  float inv = 1.f / sum;
  for (int i = lane; i < 300; i += 32) a[i] *= inv;
}

__global__ __launch_bounds__(256) void sa_av_kernel(
    const float* __restrict__ att, const float* __restrict__ v, u16* __restrict__ obf) {
  int lane = threadIdx.x & 31;
  int row  = blockIdx.x * 8 + (threadIdx.x >> 5);   // (b*8+h)*300+q
  if (row >= 4800) return;
  int q = row % 300;
  int r = row / 300;
  int h = r & 7;
  int b = r >> 3;
  const float* a  = att + (size_t)row * 300;
  const float* vp = v + (size_t)b * 300 * 256 + h * 32 + lane;
  float acc = 0.f;
  for (int k = 0; k < 300; ++k) acc += a[k] * vp[(size_t)k * 256];
  obf[((size_t)(b * 300 + q)) * 256 + h * 32 + lane] = f2bf(acc);
}

// ------------------------------ reference points ---------------------------
__global__ __launch_bounds__(256) void refpt_kernel(
    const float* __restrict__ qe, const float* __restrict__ w,
    const float* __restrict__ bb, float* __restrict__ refq) {
  int idx = blockIdx.x * 256 + threadIdx.x;
  if (idx >= 2 * 300 * 2) return;
  int comp = idx & 1;
  int q = (idx >> 1) % 300;
  float s = bb[comp];
  const float* qr = qe + (size_t)q * 512;
  for (int k = 0; k < 256; ++k) s += qr[k] * w[k * 2 + comp];
  refq[idx] = 1.f / (1.f + __expf(-s));
}

// --------------------------------- output ----------------------------------
__global__ __launch_bounds__(256) void write_out_kernel(
    const float* __restrict__ y, const float* __restrict__ refq, float* __restrict__ out) {
  int i = blockIdx.x * 256 + threadIdx.x;
  if (i >= 156000) return;
  if (i < 153600)       out[i] = y[i];
  else if (i < 154800)  out[i] = refq[i - 153600];
  else                  out[i] = refq[i - 154800];
}

// ===========================================================================
extern "C" void kernel_launch(void* const* d_in, const int* in_sizes, int n_in,
                              void* d_out, int out_size, void* d_ws, size_t ws_size,
                              hipStream_t stream) {
  (void)in_sizes; (void)n_in; (void)out_size; (void)ws_size;
  auto F = [&](int i) { return (const float*)d_in[i]; };
  // --- inputs (params pytree assumed flattened with sorted dict keys) ---
  const float* src[4] = {F(0), F(3), F(6), F(9)};
  const float* pos[4] = {F(1), F(4), F(7), F(10)};
  // dec (sorted): aw_b aw_w l1_b l1_w l2_b l2_w n1_b n1_s n2_b n2_s n3_b n3_s
  //               off_b off_w op_b op_w sa_out_b sa_out_w sa_qkv_b sa_qkv_w vp_b vp_w
  const float *d_aw_b = F(12), *d_aw_w = F(13), *d_l1_b = F(14), *d_l1_w = F(15);
  const float *d_l2_b = F(16), *d_l2_w = F(17), *d_n1_b = F(18), *d_n1_s = F(19);
  const float *d_n2_b = F(20), *d_n2_s = F(21), *d_n3_b = F(22), *d_n3_s = F(23);
  const float *d_off_b = F(24), *d_off_w = F(25), *d_op_b = F(26), *d_op_w = F(27);
  const float *d_so_b = F(28), *d_so_w = F(29), *d_qkv_b = F(30), *d_qkv_w = F(31);
  const float *d_vp_b = F(32), *d_vp_w = F(33);
  // enc (sorted): aw_b aw_w l1_b l1_w l2_b l2_w n1_b n1_s n2_b n2_s off_b off_w op_b op_w vp_b vp_w
  const float *e_aw_b = F(34), *e_aw_w = F(35), *e_l1_b = F(36), *e_l1_w = F(37);
  const float *e_l2_b = F(38), *e_l2_w = F(39), *e_n1_b = F(40), *e_n1_s = F(41);
  const float *e_n2_b = F(42), *e_n2_s = F(43), *e_off_b = F(44), *e_off_w = F(45);
  const float *e_op_b = F(46), *e_op_w = F(47), *e_vp_b = F(48), *e_vp_w = F(49);
  const float *level_embed = F(50), *query_embed = F(51);
  const float *refpt_b = F(52), *refpt_w = F(53);

  const int SB  = 26588;                 // B * total tokens
  const int SBC = SB * 256;

  // --- workspace carve-up ---
  char* wp = (char*)d_ws;
  auto alloc = [&](size_t bytes) -> void* {
    void* r = (void*)wp;
    wp += (bytes + 255) & ~(size_t)255;
    return r;
  };
  float* xbuf    = (float*)alloc((size_t)SBC * 4);
  float* posflat = (float*)alloc((size_t)SBC * 4);
  float* value   = (float*)alloc((size_t)SBC * 4);
  float* t0      = (float*)alloc((size_t)SBC * 4);
  float* offb    = (float*)alloc((size_t)SBC * 4);
  float* awb     = (float*)alloc((size_t)SB * 128 * 4);
  u16*   x_bf    = (u16*)alloc((size_t)SBC * 2);
  u16*   q_bf    = (u16*)alloc((size_t)SBC * 2);
  u16*   s_bf    = (u16*)alloc((size_t)SBC * 2);
  u16*   h_bf    = (u16*)alloc((size_t)SB * 1024 * 2);
  // packed weights (bf16 fragments); size == K*N elements each
  u16* encvp_p  = (u16*)alloc((size_t)6 * 65536 * 2);
  u16* encoff_p = (u16*)alloc((size_t)6 * 65536 * 2);
  u16* encaw_p  = (u16*)alloc((size_t)6 * 32768 * 2);
  u16* encop_p  = (u16*)alloc((size_t)6 * 65536 * 2);
  u16* encl1_p  = (u16*)alloc((size_t)6 * 262144 * 2);
  u16* encl2_p  = (u16*)alloc((size_t)6 * 262144 * 2);
  u16* decvp_p  = (u16*)alloc((size_t)6 * 65536 * 2);
  u16* decoff_p = (u16*)alloc((size_t)6 * 65536 * 2);
  u16* decaw_p  = (u16*)alloc((size_t)6 * 32768 * 2);
  u16* decop_p  = (u16*)alloc((size_t)6 * 65536 * 2);
  u16* decl1_p  = (u16*)alloc((size_t)6 * 262144 * 2);
  u16* decl2_p  = (u16*)alloc((size_t)6 * 262144 * 2);
  u16* decqk_p  = (u16*)alloc((size_t)6 * 131072 * 2);
  u16* decv_p   = (u16*)alloc((size_t)6 * 65536 * 2);
  u16* decso_p  = (u16*)alloc((size_t)6 * 65536 * 2);
  // decoder buffers (600 tokens)
  float* ybuf  = (float*)alloc(600 * 256 * 4);
  float* t0d   = (float*)alloc(600 * 256 * 4);
  float* qkd   = (float*)alloc(600 * 512 * 4);
  float* vsd   = (float*)alloc(600 * 256 * 4);
  float* attb  = (float*)alloc((size_t)4800 * 300 * 4);
  float* offd  = (float*)alloc(600 * 256 * 4);
  float* awd   = (float*)alloc(600 * 128 * 4);
  float* refq  = (float*)alloc(1200 * 4);
  u16*   y_bf  = (u16*)alloc(600 * 256 * 2);
  u16*   qd_bf = (u16*)alloc(600 * 256 * 2);
  u16*   o_bf  = (u16*)alloc(600 * 256 * 2);
  u16*   sd_bf = (u16*)alloc(600 * 256 * 2);
  u16*   hd_bf = (u16*)alloc(600 * 1024 * 2);

  auto gemm = [&](const u16* A, const u16* Bp, const float* bias, float* Cf,
                  u16* Cb, int M, int N, int K, int relu) {
    int tiles = CDIV(M, 32) * (N / 64);
    gemm_bf16_wmma<<<CDIV(tiles, 8), 256, 0, stream>>>(A, Bp, bias, Cf, Cb, M, N, K, relu);
  };
  auto pack = [&](const float* W, int K, int Nfull, int n0, int ncols, u16* dst) {
    int total = (ncols / 64) * (K / 32) * 2048;
    pack_kernel<<<CDIV(total, 256), 256, 0, stream>>>(W, dst, K, Nfull, n0, ncols);
  };

  // --- pack all weights (fp32 -> bf16 WMMA fragments) ---
  for (int i = 0; i < 6; ++i) {
    pack(e_vp_w  + (size_t)i * 65536,  256, 256,  0, 256,  encvp_p  + (size_t)i * 65536);
    pack(e_off_w + (size_t)i * 65536,  256, 256,  0, 256,  encoff_p + (size_t)i * 65536);
    pack(e_aw_w  + (size_t)i * 32768,  256, 128,  0, 128,  encaw_p  + (size_t)i * 32768);
    pack(e_op_w  + (size_t)i * 65536,  256, 256,  0, 256,  encop_p  + (size_t)i * 65536);
    pack(e_l1_w  + (size_t)i * 262144, 256, 1024, 0, 1024, encl1_p  + (size_t)i * 262144);
    pack(e_l2_w  + (size_t)i * 262144, 1024, 256, 0, 256,  encl2_p  + (size_t)i * 262144);
    pack(d_vp_w  + (size_t)i * 65536,  256, 256,  0, 256,  decvp_p  + (size_t)i * 65536);
    pack(d_off_w + (size_t)i * 65536,  256, 256,  0, 256,  decoff_p + (size_t)i * 65536);
    pack(d_aw_w  + (size_t)i * 32768,  256, 128,  0, 128,  decaw_p  + (size_t)i * 32768);
    pack(d_op_w  + (size_t)i * 65536,  256, 256,  0, 256,  decop_p  + (size_t)i * 65536);
    pack(d_l1_w  + (size_t)i * 262144, 256, 1024, 0, 1024, decl1_p  + (size_t)i * 262144);
    pack(d_l2_w  + (size_t)i * 262144, 1024, 256, 0, 256,  decl2_p  + (size_t)i * 262144);
    pack(d_qkv_w + (size_t)i * 196608, 256, 768,  0, 512,  decqk_p  + (size_t)i * 131072);
    pack(d_qkv_w + (size_t)i * 196608, 256, 768, 512, 256, decv_p   + (size_t)i * 65536);
    pack(d_so_w  + (size_t)i * 65536,  256, 256,  0, 256,  decso_p  + (size_t)i * 65536);
  }

  // --- flatten inputs + positional/level embed ---
  flatten_kernel<<<CDIV(SBC, 256), 256, 0, stream>>>(
      src[0], src[1], src[2], src[3], pos[0], pos[1], pos[2], pos[3],
      level_embed, xbuf, x_bf, posflat);

  // ============================= encoder =============================
  for (int i = 0; i < 6; ++i) {
    addq_kernel<<<CDIV(SBC, 256), 256, 0, stream>>>(xbuf, posflat, q_bf, SBC);
    gemm(x_bf, encvp_p  + (size_t)i * 65536,  e_vp_b  + i * 256, value, nullptr, SB, 256,  256, 0);
    gemm(q_bf, encoff_p + (size_t)i * 65536,  e_off_b + i * 256, offb,  nullptr, SB, 256,  256, 0);
    gemm(q_bf, encaw_p  + (size_t)i * 32768,  e_aw_b  + i * 128, awb,   nullptr, SB, 128,  256, 0);
    msda_kernel<<<SB, 256, 0, stream>>>(value, offb, awb, nullptr, s_bf, SB, 0);
    gemm(s_bf, encop_p  + (size_t)i * 65536,  e_op_b  + i * 256, t0,    nullptr, SB, 256,  256, 0);
    ln_kernel<<<CDIV(SB, 8), 256, 0, stream>>>(xbuf, t0, e_n1_s + i * 256, e_n1_b + i * 256,
                                               xbuf, x_bf, SB);
    gemm(x_bf, encl1_p + (size_t)i * 262144, e_l1_b + i * 1024, nullptr, h_bf, SB, 1024, 256, 1);
    gemm(h_bf, encl2_p + (size_t)i * 262144, e_l2_b + i * 256,  t0, nullptr,   SB, 256, 1024, 0);
    ln_kernel<<<CDIV(SB, 8), 256, 0, stream>>>(xbuf, t0, e_n2_s + i * 256, e_n2_b + i * 256,
                                               xbuf, x_bf, SB);
  }
  // xbuf / x_bf now hold encoder memory.

  // ============================= decoder =============================
  dec_init_kernel<<<CDIV(600 * 256, 256), 256, 0, stream>>>(query_embed, ybuf, y_bf);
  refpt_kernel<<<CDIV(1200, 256), 256, 0, stream>>>(query_embed, refpt_w, refpt_b, refq);

  for (int i = 0; i < 6; ++i) {
    // ---- self attention ----
    dec_addq_kernel<<<CDIV(600 * 256, 256), 256, 0, stream>>>(ybuf, query_embed, qd_bf);
    gemm(qd_bf, decqk_p + (size_t)i * 131072, d_qkv_b + i * 768,       qkd, nullptr, 600, 512, 256, 0);
    gemm(y_bf,  decv_p  + (size_t)i * 65536,  d_qkv_b + i * 768 + 512, vsd, nullptr, 600, 256, 256, 0);
    sa_scores_kernel<<<CDIV(2 * 8 * 300 * 300, 256), 256, 0, stream>>>(qkd, attb);
    sa_softmax_kernel<<<CDIV(4800, 8), 256, 0, stream>>>(attb);
    sa_av_kernel<<<CDIV(4800, 8), 256, 0, stream>>>(attb, vsd, o_bf);
    gemm(o_bf, decso_p + (size_t)i * 65536, d_so_b + i * 256, t0d, nullptr, 600, 256, 256, 0);
    ln_kernel<<<CDIV(600, 8), 256, 0, stream>>>(ybuf, t0d, d_n2_s + i * 256, d_n2_b + i * 256,
                                                ybuf, y_bf, 600);
    // ---- cross deformable attention over memory ----
    dec_addq_kernel<<<CDIV(600 * 256, 256), 256, 0, stream>>>(ybuf, query_embed, qd_bf);
    gemm(x_bf,  decvp_p  + (size_t)i * 65536, d_vp_b  + i * 256, value, nullptr, SB, 256, 256, 0);
    gemm(qd_bf, decoff_p + (size_t)i * 65536, d_off_b + i * 256, offd,  nullptr, 600, 256, 256, 0);
    gemm(qd_bf, decaw_p  + (size_t)i * 32768, d_aw_b  + i * 128, awd,   nullptr, 600, 128, 256, 0);
    msda_kernel<<<600, 256, 0, stream>>>(value, offd, awd, refq, sd_bf, 600, 1);
    gemm(sd_bf, decop_p + (size_t)i * 65536, d_op_b + i * 256, t0d, nullptr, 600, 256, 256, 0);
    ln_kernel<<<CDIV(600, 8), 256, 0, stream>>>(ybuf, t0d, d_n1_s + i * 256, d_n1_b + i * 256,
                                                ybuf, y_bf, 600);
    // ---- FFN ----
    gemm(y_bf,  decl1_p + (size_t)i * 262144, d_l1_b + i * 1024, nullptr, hd_bf, 600, 1024, 256, 1);
    gemm(hd_bf, decl2_p + (size_t)i * 262144, d_l2_b + i * 256,  t0d, nullptr,   600, 256, 1024, 0);
    ln_kernel<<<CDIV(600, 8), 256, 0, stream>>>(ybuf, t0d, d_n3_s + i * 256, d_n3_b + i * 256,
                                                ybuf, y_bf, 600);
  }

  write_out_kernel<<<CDIV(156000, 256), 256, 0, stream>>>(ybuf, refq, (float*)d_out);
}
